// MHA_39848706573825
// MI455X (gfx1250) — matrix operand order repaired and checked by reading, not compiled, not run on hardware
//
#include <hip/hip_runtime.h>
#include <hip/hip_bf16.h>
#include <math.h>

// ---------------------------------------------------------------------------
// MHA forward for B=2, T=2048, D=1024, H=16, HD=64 on gfx1250 (CDNA5, wave32)
// GEMMs on v_wmma_f32_16x16x32_f16; tile staging via global_load_async_to_lds
// (ASYNCcnt); cross-lane softmax reductions via ds_swizzle XOR patterns.
// ---------------------------------------------------------------------------

typedef _Float16 half_t;
typedef __attribute__((ext_vector_type(16))) _Float16 v16h;
typedef __attribute__((ext_vector_type(8)))  _Float16 v8h;
typedef __attribute__((ext_vector_type(8)))  float    v8f;

union H16 { v16h v16; v8h v8[2]; half_t h[16]; };
union PK2 { half_t h[2]; uint32_t u; };

#define BB 2
#define TT 2048
#define DD 1024
#define HH 16
#define HD 64

static __device__ __forceinline__ v8f wmma_f16(v16h a, v16h b, v8f c) {
  return __builtin_amdgcn_wmma_f32_16x16x32_f16(
      false, a, false, b, (short)0, c, false, false);
}

// low 32 bits of a flat pointer to LDS == LDS byte address
static __device__ __forceinline__ uint32_t lds_addr32(const void* p) {
  return (uint32_t)(uintptr_t)p;
}

// async DMA: 16 bytes global -> LDS, same immediate offset on both sides
#define ASYNC_B128(ldsa, gp, off_)                                          \
  asm volatile("global_load_async_to_lds_b128 %0, %1, off offset:" #off_    \
               :: "v"(ldsa), "v"(gp) : "memory")
#define WAIT_ASYNC() asm volatile("s_wait_asynccnt 0x0" ::: "memory")

// cross-lane XOR exchange within 32 lanes via ds_swizzle (1 DS op, no VALU)
template <int XORM>
static __device__ __forceinline__ float swz_xor(float v) {
  return __int_as_float(__builtin_amdgcn_ds_swizzle(
      __float_as_int(v), (XORM << 10) | 0x1f));
}

// ---------------------------------------------------------------------------
__global__ __launch_bounds__(256) void f32_to_f16_kernel(
    const float* __restrict__ in, half_t* __restrict__ out, int n) {
  int i = blockIdx.x * 256 + threadIdx.x;
  if (i < n) out[i] = (half_t)in[i];
}

// ---------------------------------------------------------------------------
// One K-step (K=32) of the 128x128 workgroup tile: 8 WMMAs per wave.
// ---------------------------------------------------------------------------
static __device__ __forceinline__ void gemm_step(
    const half_t* sAbuf, const half_t* sBbuf, int wm, int wn,
    int m, int kh, int koff, v8f (&acc)[2][4]) {
  v16h af[2], bf[4];
#pragma unroll
  for (int fm = 0; fm < 2; fm++) {
    const half_t* p = sAbuf + (wm * 32 + fm * 16 + m) * 40;
    H16 u;
    u.v8[0] = *(const v8h*)(p + kh);
    u.v8[1] = *(const v8h*)(p + kh + 16);
    af[fm] = u.v16;
  }
#pragma unroll
  for (int fn = 0; fn < 4; fn++) {
    const half_t* p = sBbuf + (wn * 64 + fn * 16 + m) * 40 + kh;
    H16 u;
    u.v8[0] = *(const v8h*)(p);
    u.v8[1] = *(const v8h*)(p + 16);
    bf[fn] = u.v16;
  }
#pragma unroll
  for (int fm = 0; fm < 2; fm++)
#pragma unroll
    for (int fn = 0; fn < 4; fn++)
      acc[fm][fn] = wmma_f16(af[fm], bf[fn], acc[fm][fn]);
}

// ---------------------------------------------------------------------------
// C[M,N] = A[M,K] * B[N,K]^T, f16 in, f32 accum. 128x128 workgroup tile,
// LDS double-buffered, staged with global_load_async_to_lds_b128.
// ---------------------------------------------------------------------------
template <bool F16OUT>
__global__ __launch_bounds__(256) void gemm_xwT(
    const half_t* __restrict__ A, const half_t* __restrict__ Bw,
    void* __restrict__ Cout, int M, int N, int K) {
  __shared__ half_t sA[2][128 * 40];  // 128 rows x 32 halves, stride 40
  __shared__ half_t sB[2][128 * 40];

  const int tid  = threadIdx.x;
  const int lane = tid & 31;
  const int wave = tid >> 5;
  const int wm   = wave >> 1;
  const int wn   = wave & 1;
  const int m0   = blockIdx.y * 128;
  const int n0   = blockIdx.x * 128;

  const v8f vzero = {0.f, 0.f, 0.f, 0.f, 0.f, 0.f, 0.f, 0.f};
  v8f acc[2][4];
#pragma unroll
  for (int i = 0; i < 2; i++)
#pragma unroll
    for (int j = 0; j < 4; j++) acc[i][j] = vzero;

  const int ldrow  = tid >> 1;         // 0..127
  const int ldpart = (tid & 1) * 16;   // 0 or 16 halves (32B per thread)
  const half_t* gA = A  + (size_t)(m0 + ldrow) * K + ldpart;
  const half_t* gB = Bw + (size_t)(n0 + ldrow) * K + ldpart;
  const int lds_o  = ldrow * 40 + ldpart;
  const uint32_t lA[2] = {lds_addr32(sA[0] + lds_o), lds_addr32(sA[1] + lds_o)};
  const uint32_t lB[2] = {lds_addr32(sB[0] + lds_o), lds_addr32(sB[1] + lds_o)};

  const int m    = lane & 15;
  const int kh   = (lane >> 4) * 8;
  const int koff = (lane >> 4) * 16;

  // prologue: k=0 into buffer 0
  ASYNC_B128(lA[0], gA, 0);
  ASYNC_B128(lA[0], gA, 16);
  ASYNC_B128(lB[0], gB, 0);
  ASYNC_B128(lB[0], gB, 16);
  WAIT_ASYNC();

  int cur = 0;
  for (int k0 = 32; k0 < K; k0 += 32) {
    __syncthreads();                     // publish buf[cur] from all waves
    const half_t* ga = gA + k0;
    const half_t* gb = gB + k0;
    ASYNC_B128(lA[cur ^ 1], ga, 0);      // stage next while computing
    ASYNC_B128(lA[cur ^ 1], ga, 16);
    ASYNC_B128(lB[cur ^ 1], gb, 0);
    ASYNC_B128(lB[cur ^ 1], gb, 16);
    gemm_step(sA[cur], sB[cur], wm, wn, m, kh, koff, acc);
    WAIT_ASYNC();                        // own DMA done before next barrier
    cur ^= 1;
  }
  __syncthreads();
  gemm_step(sA[cur], sB[cur], wm, wn, m, kh, koff, acc);

  const int rowoff = 8 * (lane >> 4);
#pragma unroll
  for (int fm = 0; fm < 2; fm++)
#pragma unroll
    for (int fn = 0; fn < 4; fn++)
#pragma unroll
      for (int r = 0; r < 8; r++) {
        int row = m0 + wm * 32 + fm * 16 + r + rowoff;
        int col = n0 + wn * 64 + fn * 16 + m;
        if (F16OUT)
          ((half_t*)Cout)[(size_t)row * N + col] = (half_t)acc[fm][fn][r];
        else
          ((float*)Cout)[(size_t)row * N + col] = acc[fm][fn][r];
      }
}

// ---------------------------------------------------------------------------
// RoPE in place on q,k; 1/sqrt(HD)=0.125 folded into q (exact in f16).
// ---------------------------------------------------------------------------
__global__ __launch_bounds__(256) void rope_kernel(
    half_t* __restrict__ qkv, const float* __restrict__ rot) {
  int i  = blockIdx.x * 256 + threadIdx.x;
  int hd = i & 31;
  int h  = (i >> 5) & 15;
  int s  = (i >> 9) & 1;
  int t  = (i >> 10) & (TT - 1);
  int b  = i >> 21;
  size_t base = ((size_t)(b * TT + t)) * (3 * DD) + s * DD + h * HD;
  float a0 = rot[t * HD + hd];
  float a1 = rot[t * HD + hd + 32];
  float x0 = (float)qkv[base + hd];
  float x1 = (float)qkv[base + hd + 32];
  float scale = (s == 0) ? 0.125f : 1.0f;
  qkv[base + hd]      = (half_t)((x0 * __cosf(a0) - x1 * __sinf(a0)) * scale);
  qkv[base + hd + 32] = (half_t)((x1 * __cosf(a1) + x0 * __sinf(a1)) * scale);
}

// ---------------------------------------------------------------------------
// Causal flash attention. Waves 4..7 stage the K tile via async DMA;
// waves 0..3 stage V transposed (b32-packed pairs of keys). Double-buffered.
// ---------------------------------------------------------------------------
__global__ __launch_bounds__(256) void flash_attn(
    const half_t* __restrict__ qkv, half_t* __restrict__ attn_out) {
  __shared__ half_t sK[2][64 * 72];     // [key][hd]
  __shared__ half_t sV[2][64 * 72];     // transposed: [hd][key]
  __shared__ half_t sP[8 * 16 * 72];    // per-wave P scratch

  const int tid  = threadIdx.x;
  const int lane = tid & 31;
  const int wave = tid >> 5;
  const int q0   = blockIdx.x * 128;
  const int h    = blockIdx.y;
  const int b    = blockIdx.z;
  const int qrow0 = q0 + wave * 16;

  const size_t rowstride = 3 * DD;
  const half_t* Qbase = qkv + (size_t)b * TT * rowstride + h * HD;
  const half_t* Kbase = Qbase + DD;
  const half_t* Vbase = Qbase + 2 * DD;

  const int m      = lane & 15;
  const int kh     = (lane >> 4) * 8;
  const int koff   = (lane >> 4) * 16;
  const int rowoff = 8 * (lane >> 4);

  v16h qa[2];
  {
    const half_t* p = Qbase + (size_t)(qrow0 + m) * rowstride;
#pragma unroll
    for (int ks = 0; ks < 2; ks++) {
      H16 u;
      u.v8[0] = *(const v8h*)(p + ks * 32 + kh);
      u.v8[1] = *(const v8h*)(p + ks * 32 + kh + 16);
      qa[ks] = u.v16;
    }
  }

  const v8f vzero = {0.f, 0.f, 0.f, 0.f, 0.f, 0.f, 0.f, 0.f};
  float mrun[8], lsum[8];
  v8f oacc[4];
#pragma unroll
  for (int r = 0; r < 8; r++) { mrun[r] = -1e30f; lsum[r] = 0.f; }
#pragma unroll
  for (int nf = 0; nf < 4; nf++) oacc[nf] = vzero;

  const int nblocks = blockIdx.x * 2 + 2;
  half_t* sPw = sP + wave * 16 * 72;

  // staging roles (uniform per wave)
  const bool kstager = (tid >= 128);
  const int  ku  = tid & 127;
  const int  kk  = ku >> 1;            // key row 0..63
  const int  kc  = (ku & 1) * 32;      // 64B half of the row
  const half_t* gK = Kbase + (size_t)kk * rowstride + kc;
  const uint32_t lK[2] = {lds_addr32(sK[0] + kk * 72 + kc),
                          lds_addr32(sK[1] + kk * 72 + kc)};
  const int  vp  = ku >> 2;            // key pair 0..31
  const int  vc  = (ku & 3) * 16;      // hd chunk
  const half_t* gV0 = Vbase + (size_t)(2 * vp) * rowstride + vc;

  H16 uv0, uv1;
  if (kstager) {
    ASYNC_B128(lK[0], gK, 0);
    ASYNC_B128(lK[0], gK, 16);
    ASYNC_B128(lK[0], gK, 32);
    ASYNC_B128(lK[0], gK, 48);
  } else {
    uv0.v16 = *(const v16h*)(gV0);
    uv1.v16 = *(const v16h*)(gV0 + rowstride);
#pragma unroll
    for (int j = 0; j < 16; j++) {
      PK2 pk; pk.h[0] = uv0.h[j]; pk.h[1] = uv1.h[j];
      *(uint32_t*)(sV[0] + (vc + j) * 72 + 2 * vp) = pk.u;
    }
  }
  WAIT_ASYNC();

  int cur = 0;
  for (int kb = 0; kb < nblocks; kb++) {
    const int key0 = kb * 64;
    const bool haveNext = (kb + 1 < nblocks);
    if (haveNext && !kstager) {  // V loads for next block before the barrier
      const half_t* g = gV0 + (size_t)(key0 + 64) * rowstride;
      uv0.v16 = *(const v16h*)(g);
      uv1.v16 = *(const v16h*)(g + rowstride);
    }
    __syncthreads();
    if (haveNext) {
      if (kstager) {
        const half_t* g = gK + (size_t)(key0 + 64) * rowstride;
        ASYNC_B128(lK[cur ^ 1], g, 0);
        ASYNC_B128(lK[cur ^ 1], g, 16);
        ASYNC_B128(lK[cur ^ 1], g, 32);
        ASYNC_B128(lK[cur ^ 1], g, 48);
      } else {
        half_t* dV = sV[cur ^ 1];
#pragma unroll
        for (int j = 0; j < 16; j++) {
          PK2 pk; pk.h[0] = uv0.h[j]; pk.h[1] = uv1.h[j];
          *(uint32_t*)(dV + (vc + j) * 72 + 2 * vp) = pk.u;
        }
      }
    }

    if (key0 <= qrow0 + 15) {  // causal wave-level block skip
      v8f sfr[4];
#pragma unroll
      for (int nf = 0; nf < 4; nf++) {
        v8f c = vzero;
#pragma unroll
        for (int ks = 0; ks < 2; ks++) {
          const half_t* p = sK[cur] + (nf * 16 + m) * 72 + ks * 32 + kh;
          H16 u;
          u.v8[0] = *(const v8h*)(p);
          u.v8[1] = *(const v8h*)(p + 16);
          c = wmma_f16(qa[ks], u.v16, c);
        }
        sfr[nf] = c;
      }

      const bool needmask = (key0 + 63) > qrow0;
#pragma unroll
      for (int r = 0; r < 8; r++) {
        const int qrow = qrow0 + r + rowoff;
        float mx = -1e30f;
#pragma unroll
        for (int nf = 0; nf < 4; nf++) {
          float s = sfr[nf][r];
          if (needmask && (key0 + nf * 16 + m) > qrow) s = -1e30f;
          sfr[nf][r] = s;
          mx = fmaxf(mx, s);
        }
        mx = fmaxf(mx, swz_xor<1>(mx));
        mx = fmaxf(mx, swz_xor<2>(mx));
        mx = fmaxf(mx, swz_xor<4>(mx));
        mx = fmaxf(mx, swz_xor<8>(mx));
        float mnew = fmaxf(mrun[r], mx);
        float alpha = __expf(mrun[r] - mnew);
        mrun[r] = mnew;
        float psum = 0.f;
#pragma unroll
        for (int nf = 0; nf < 4; nf++) {
          float p = __expf(sfr[nf][r] - mnew);
          sfr[nf][r] = p;
          psum += p;
        }
        psum += swz_xor<1>(psum);
        psum += swz_xor<2>(psum);
        psum += swz_xor<4>(psum);
        psum += swz_xor<8>(psum);
        lsum[r] = lsum[r] * alpha + psum;
#pragma unroll
        for (int nf = 0; nf < 4; nf++) oacc[nf][r] *= alpha;
      }

#pragma unroll
      for (int nf = 0; nf < 4; nf++)
#pragma unroll
        for (int r = 0; r < 8; r++)
          sPw[(r + rowoff) * 72 + nf * 16 + m] = (half_t)sfr[nf][r];

      v16h pa[2];
#pragma unroll
      for (int ks = 0; ks < 2; ks++) {
        const half_t* p = sPw + m * 72 + ks * 32;
        H16 u;
        u.v8[0] = *(const v8h*)(p + kh);
        u.v8[1] = *(const v8h*)(p + kh + 16);
        pa[ks] = u.v16;
      }
#pragma unroll
      for (int nf = 0; nf < 4; nf++) {
#pragma unroll
        for (int ks = 0; ks < 2; ks++) {
          const half_t* p = sV[cur] + (nf * 16 + m) * 72 + ks * 32 + kh;
          H16 u;
          u.v8[0] = *(const v8h*)(p);
          u.v8[1] = *(const v8h*)(p + 16);
          oacc[nf] = wmma_f16(pa[ks], u.v16, oacc[nf]);
        }
      }
    }
    WAIT_ASYNC();  // own K-tile DMA done before the next barrier
    cur ^= 1;
  }

#pragma unroll
  for (int nf = 0; nf < 4; nf++)
#pragma unroll
    for (int r = 0; r < 8; r++) {
      int trow = qrow0 + r + rowoff;
      attn_out[(size_t)(b * TT + trow) * DD + h * HD + nf * 16 + m] =
          (half_t)(oacc[nf][r] / lsum[r]);
    }
}

// ---------------------------------------------------------------------------
extern "C" void kernel_launch(void* const* d_in, const int* in_sizes, int n_in,
                              void* d_out, int out_size, void* d_ws, size_t ws_size,
                              hipStream_t stream) {
  const float* x    = (const float*)d_in[0];
  const float* rot  = (const float*)d_in[1];
  const float* wqkv = (const float*)d_in[2];
  const float* wout = (const float*)d_in[3];

  const size_t nx   = (size_t)BB * TT * DD;
  const size_t nwq  = (size_t)3 * DD * DD;
  const size_t nwo  = (size_t)DD * DD;
  const size_t nqkv = (size_t)BB * TT * 3 * DD;
  const size_t natt = (size_t)BB * TT * DD;

  char* ws = (char*)d_ws;
  half_t* xh   = (half_t*)ws;            ws += nx * sizeof(half_t);
  half_t* wqh  = (half_t*)ws;            ws += nwq * sizeof(half_t);
  half_t* woh  = (half_t*)ws;            ws += nwo * sizeof(half_t);
  half_t* qkvh = (half_t*)ws;            ws += nqkv * sizeof(half_t);
  half_t* atth = (half_t*)ws;            ws += natt * sizeof(half_t);

  f32_to_f16_kernel<<<(int)(nx / 256), 256, 0, stream>>>(x, xh, (int)nx);
  f32_to_f16_kernel<<<(int)(nwq / 256), 256, 0, stream>>>(wqkv, wqh, (int)nwq);
  f32_to_f16_kernel<<<(int)(nwo / 256), 256, 0, stream>>>(wout, woh, (int)nwo);

  gemm_xwT<true><<<dim3(3 * DD / 128, BB * TT / 128), 256, 0, stream>>>(
      xh, wqh, qkvh, BB * TT, 3 * DD, DD);

  rope_kernel<<<(BB * TT * 2 * HH * 32) / 256, 256, 0, stream>>>(qkvh, rot);

  flash_attn<<<dim3(TT / 128, HH, BB), 256, 0, stream>>>(qkvh, atth);

  gemm_xwT<false><<<dim3(DD / 128, BB * TT / 128), 256, 0, stream>>>(
      atth, woh, d_out, BB * TT, DD, DD);
}
